// BRIOSImputer_66322884985486
// MI455X (gfx1250) — compile-verified, matrix operand order checked by reading, not run on hardware
//
#include <hip/hip_runtime.h>
#include <math.h>

typedef __attribute__((ext_vector_type(16))) _Float16 v16h;
typedef __attribute__((ext_vector_type(4)))  _Float16 v4h;
typedef __attribute__((ext_vector_type(8)))  float    v8f;

#define B_   512
#define L_   1024
#define D_   16
#define H_   64
#define BT   16    // batch tile (= WMMA M)
#define TPB  128   // 4 waves of 32
#define CH   64    // steps per staged chunk (x/dt in, y out)

// K index inside a 16-bit A/B WMMA fragment (16x32 / 32x16), per ISA 7.12.2:
// lanes 0-15 hold K=0..7 (v0..3) and K=16..23 (v4..7);
// lanes 16-31 hold K=8..15 and K=24..31. Two f16 per VGPR (idx = 2v+p).
__device__ __forceinline__ int kmap(int lane, int idx) {
    int v = idx >> 1, p = idx & 1;
    return ((lane >> 4) & 1) * 8 + ((v >> 2) & 1) * 16 + ((v & 3) << 1) + p;
}

// fast sigmoid: v_exp_f32 + v_rcp_f32 (no IEEE divide expansion)
__device__ __forceinline__ float fsig(float xv) {
    return __builtin_amdgcn_rcpf(1.0f + __expf(-xv));
}

__global__ __launch_bounds__(TPB) void brios_gru_kernel(
    const float* __restrict__ x,     const float* __restrict__ dt,
    const float* __restrict__ f_Wih, const float* __restrict__ f_Whh,
    const float* __restrict__ f_bih, const float* __restrict__ f_bhh,
    const float* __restrict__ f_gamma, const float* __restrict__ f_Wout,
    const float* __restrict__ f_bout,
    const float* __restrict__ b_Wih, const float* __restrict__ b_Whh,
    const float* __restrict__ b_bih, const float* __restrict__ b_bhh,
    const float* __restrict__ b_gamma, const float* __restrict__ b_Wout,
    const float* __restrict__ b_bout,
    float* __restrict__ out)   // [ y | y_f | y_b ], each B_*L_
{
    const int wgPerDir = B_ / BT;            // 32
    const int dir = blockIdx.x / wgPerDir;   // 0 = forward, 1 = backward
    const int wb  = (blockIdx.x % wgPerDir) * BT;

    const float* Wih  = dir ? b_Wih  : f_Wih;
    const float* Whh  = dir ? b_Whh  : f_Whh;
    const float* bih  = dir ? b_bih  : f_bih;
    const float* bhh  = dir ? b_bhh  : f_bhh;
    const float* Wout = dir ? b_Wout : f_Wout;
    const float  gamma = dir ? b_gamma[0] : f_gamma[0];
    const float  bout  = dir ? b_bout[0]  : f_bout[0];
    float* ydir = out + (size_t)(dir ? 2 : 1) * (B_ * L_);

    const int tid  = threadIdx.x;
    const int lane = tid & 31;
    const int wave = tid >> 5;

    __shared__ _Float16 hb16[BT * H_];            // h_bar f16 for current step
    __shared__ _Float16 xchunk[BT * CH * D_];     // x staged for CH steps (f16)
    __shared__ float    dtchunk[BT * CH];         // dt_{s+1} staged, shifted
    __shared__ float    grz[BT * 128];            // fused gi+gh+biases for r,z
    __shared__ float    gin[BT * 64];             // gi for n gate (+bih_n)
    __shared__ float    ghn[BT * 64];             // gh for n gate (+bhh_n)
    __shared__ float    ybuf[BT * CH];            // buffered y outputs

    for (int i = tid; i < BT * H_; i += TPB) hb16[i] = (_Float16)0.0f;

    const float g = fminf(fmaxf(gamma, 1e-4f), 10.0f);

    // ---- Time-invariant per-lane state: B fragments + bias-initialized C ----
    const int nc = lane & 15;
    v16h Brz_ih[2], Brz_hh[2][2];   // r/z tiles (fused gi+gh accumulation)
    v16h Bn_ih, Bn_hh[2];           // n tile (gi, gh kept separate)
    v8f  c_rz[2], c_bi, c_bh;       // persistent bias-initialized C operands
    #pragma unroll
    for (int tI = 0; tI < 2; ++tI) {
        const int row = wave * 32 + tI * 16 + nc;          // rows 0..127
        #pragma unroll
        for (int idx = 0; idx < 16; ++idx) {
            const int k = kmap(lane, idx);
            Brz_ih[tI][idx]    = (idx < 8) ? (_Float16)Wih[row * D_ + k]
                                           : (_Float16)0.0f;
            Brz_hh[tI][0][idx] = (_Float16)Whh[row * H_ + k];
            Brz_hh[tI][1][idx] = (_Float16)Whh[row * H_ + 32 + k];
        }
        const float bsum = bih[row] + bhh[row];
        #pragma unroll
        for (int v = 0; v < 8; ++v) c_rz[tI][v] = bsum;
    }
    {
        const int row = 128 + wave * 16 + nc;              // n-gate rows
        #pragma unroll
        for (int idx = 0; idx < 16; ++idx) {
            const int k = kmap(lane, idx);
            Bn_ih[idx]    = (idx < 8) ? (_Float16)Wih[row * D_ + k]
                                      : (_Float16)0.0f;
            Bn_hh[0][idx] = (_Float16)Whh[row * H_ + k];
            Bn_hh[1][idx] = (_Float16)Whh[row * H_ + 32 + k];
        }
        const float bi = bih[row], bh = bhh[row];
        #pragma unroll
        for (int v = 0; v < 8; ++v) { c_bi[v] = bi; c_bh[v] = bh; }
    }

    // per-thread pointwise geometry (time-invariant)
    const int b  = tid >> 3;          // batch row this thread owns
    const int j0 = (tid & 7) << 3;    // first of 8 hidden indices
    float wo[8];
    #pragma unroll
    for (int q = 0; q < 8; ++q) wo[q] = Wout[j0 + q];
    float hbar[8];                    // h_bar for current step, f32 registers
    #pragma unroll
    for (int q = 0; q < 8; ++q) hbar[q] = 0.0f;

    const int m      = lane & 15;          // WMMA A/D row
    const int mrowHi = (lane >> 4) << 3;   // D rows: v + {0,8}

    // Ax upper half (K>=16) is always zero; set once, overwrite low half per step
    v16h Ax;
    #pragma unroll
    for (int idx = 8; idx < 16; ++idx) Ax[idx] = (_Float16)0.0f;

    __syncthreads();

    for (int s = 0; s < L_; ++s) {
        const int c = s & (CH - 1);

        if (c == 0) {
            // ---- stage x[steps s..s+CH) and dt[steps s+1..s+CH] into LDS ----
            const int s0 = s;
            const int tlo = dir ? (L_ - CH - s0) : s0;   // memory-contiguous base
            for (int i4 = tid; i4 < (BT * CH * D_) / 4; i4 += TPB) {
                const int idx = i4 << 2;
                const int bb  = idx >> 10;          // / (CH*D_)
                const int rem = idx & 1023;
                const int u   = rem >> 4;           // t offset in memory order
                const int d   = rem & 15;
                const int cc  = dir ? (CH - 1 - u) : u;
                const float4 xv = *(const float4*)&x[((size_t)(wb + bb) * L_ + (tlo + u)) * D_ + d];
                v4h hv;
                hv[0] = (_Float16)xv.x; hv[1] = (_Float16)xv.y;
                hv[2] = (_Float16)xv.z; hv[3] = (_Float16)xv.w;
                *(v4h*)&xchunk[bb * (CH * D_) + cc * D_ + d] = hv;
            }
            for (int i = tid; i < BT * CH; i += TPB) {
                const int bb = i >> 6;
                const int u  = i & (CH - 1);
                int tg = dir ? (L_ - 2 - (s0 + u)) : (s0 + 1 + u);
                tg = max(0, min(L_ - 1, tg));       // clamp (only last step, unused)
                dtchunk[bb * CH + u] = dt[(size_t)(wb + bb) * L_ + tg];
            }
            __syncthreads();
        }

        // ---- phase B: WMMA ----
        v16h Ah0, Ah1;
        const _Float16* xrow = &xchunk[m * (CH * D_) + c * D_];
        #pragma unroll
        for (int idx = 0; idx < 16; ++idx) {
            const int k = kmap(lane, idx);
            Ah0[idx] = hb16[m * H_ + k];
            Ah1[idx] = hb16[m * H_ + 32 + k];
            if (idx < 8) Ax[idx] = xrow[k];
        }
        #pragma unroll
        for (int tI = 0; tI < 2; ++tI) {   // fused r/z tiles
            v8f acc;
            acc = __builtin_amdgcn_wmma_f32_16x16x32_f16(false, Ax,  false, Brz_ih[tI],
                                                         (short)0, c_rz[tI], false, false);
            acc = __builtin_amdgcn_wmma_f32_16x16x32_f16(false, Ah0, false, Brz_hh[tI][0],
                                                         (short)0, acc, false, false);
            acc = __builtin_amdgcn_wmma_f32_16x16x32_f16(false, Ah1, false, Brz_hh[tI][1],
                                                         (short)0, acc, false, false);
            const int nG = wave * 32 + tI * 16 + nc;
            #pragma unroll
            for (int v = 0; v < 8; ++v) grz[(v + mrowHi) * 128 + nG] = acc[v];
        }
        {   // n-gate tile: D stored straight to LDS (no repacking movs/hazard nops)
            v8f ci, ch;
            ci = __builtin_amdgcn_wmma_f32_16x16x32_f16(false, Ax,  false, Bn_ih,
                                                        (short)0, c_bi, false, false);
            ch = __builtin_amdgcn_wmma_f32_16x16x32_f16(false, Ah0, false, Bn_hh[0],
                                                        (short)0, c_bh, false, false);
            ch = __builtin_amdgcn_wmma_f32_16x16x32_f16(false, Ah1, false, Bn_hh[1],
                                                        (short)0, ch, false, false);
            const int nGn = wave * 16 + nc;
            #pragma unroll
            for (int v = 0; v < 8; ++v) {
                gin[(v + mrowHi) * 64 + nGn] = ci[v];
                ghn[(v + mrowHi) * 64 + nGn] = ch[v];
            }
        }
        __syncthreads();

        // ---- phase C: gates, h update (registers), y reduce, next-step prep ----
        float ypart = 0.0f;
        float hnew[8];
        #pragma unroll
        for (int q = 0; q < 8; ++q) {
            const int j = j0 + q;
            const float r = fsig(grz[b * 128 + j]);
            const float z = fsig(grz[b * 128 + 64 + j]);
            const float n = fmaf(2.0f, fsig(2.0f * fmaf(r, ghn[b * 64 + j], gin[b * 64 + j])), -1.0f);
            const float hv = (1.0f - z) * n + z * hbar[q];
            hnew[q] = hv;
            ypart += hv * wo[q];
        }
        // 8-lane in-wave reduction (partners are consecutive lanes)
        ypart += __shfl_xor(ypart, 1, 32);
        ypart += __shfl_xor(ypart, 2, 32);
        ypart += __shfl_xor(ypart, 4, 32);
        if ((tid & 7) == 0) ybuf[b * CH + c] = bout + ypart;

        // prepare next step: h_bar_{s+1} = h_s * decay(dt_{s+1})
        {
            float dd = dtchunk[b * CH + c];
            dd = fminf(fmaxf(dd, 0.0f), 1.0e6f);
            const float decay = __expf(-g * dd);
            #pragma unroll
            for (int q = 0; q < 8; ++q) {
                const float hb = hnew[q] * decay;
                hbar[q] = hb;
                hb16[b * H_ + j0 + q] = (_Float16)hb;
            }
        }
        __syncthreads();

        if (c == CH - 1) {   // coalesced y flush every CH steps
            const int s0 = s - (CH - 1);
            for (int i = tid; i < BT * CH; i += TPB) {
                const int bb = i >> 6, cc = i & (CH - 1);
                const int tt = dir ? (L_ - 1 - (s0 + cc)) : (s0 + cc);
                ydir[(size_t)(wb + bb) * L_ + tt] = ybuf[i];
            }
            __syncthreads();
        }
    }
}

__global__ void combine_y_kernel(float* __restrict__ out) {
    const size_t N = (size_t)B_ * L_;
    const size_t i = (size_t)blockIdx.x * blockDim.x + threadIdx.x;
    if (i < N) out[i] = 0.5f * (out[N + i] + out[2 * N + i]);
}

extern "C" void kernel_launch(void* const* d_in, const int* in_sizes, int n_in,
                              void* d_out, int out_size, void* d_ws, size_t ws_size,
                              hipStream_t stream) {
    (void)in_sizes; (void)n_in; (void)d_ws; (void)ws_size; (void)out_size;
    const float* x      = (const float*)d_in[0];
    const float* dt     = (const float*)d_in[1];
    const float* f_Wih  = (const float*)d_in[2];
    const float* f_Whh  = (const float*)d_in[3];
    const float* f_bih  = (const float*)d_in[4];
    const float* f_bhh  = (const float*)d_in[5];
    const float* f_gam  = (const float*)d_in[6];
    const float* f_Wout = (const float*)d_in[7];
    const float* f_bout = (const float*)d_in[8];
    const float* b_Wih  = (const float*)d_in[9];
    const float* b_Whh  = (const float*)d_in[10];
    const float* b_bih  = (const float*)d_in[11];
    const float* b_bhh  = (const float*)d_in[12];
    const float* b_gam  = (const float*)d_in[13];
    const float* b_Wout = (const float*)d_in[14];
    const float* b_bout = (const float*)d_in[15];
    float* out = (float*)d_out;

    brios_gru_kernel<<<dim3(2 * (B_ / BT)), dim3(TPB), 0, stream>>>(
        x, dt, f_Wih, f_Whh, f_bih, f_bhh, f_gam, f_Wout, f_bout,
        b_Wih, b_Whh, b_bih, b_bhh, b_gam, b_Wout, b_bout, out);

    const int n = B_ * L_;
    combine_y_kernel<<<dim3((n + 255) / 256), dim3(256), 0, stream>>>(out);
}
